// decoder_82377472737758
// MI455X (gfx1250) — matrix operand order repaired and checked by reading, compiled, not run on hardware
//
#include <hip/hip_runtime.h>
#include <math.h>

#define TSTEPS 2048
#define DTC    0.001f
#define D0     256
#define D1     512
#define D2     1024
#define KMIX   16
#define NPIX   128
#define NWG    64
#define NTHR   256
#define NWAVES (NWG * (NTHR / 32))   // 512 waves (wave32)

typedef __attribute__((ext_vector_type(16))) __bf16    v16bf;
typedef __attribute__((ext_vector_type(8)))  float     v8f;
typedef __attribute__((ext_vector_type(4)))  unsigned  v4u;

union BFPack { v16bf v; v4u q[2]; };

// ---- output layout (return-order concatenation, all fp32) ----
constexpr size_t O_A0   = 0;
constexpr size_t O_A1o  = O_A0  + (size_t)TSTEPS * 16;
constexpr size_t O_A2o  = O_A1o + (size_t)TSTEPS * 16;
constexpr size_t O_X0   = O_A2o + (size_t)TSTEPS * 16;
constexpr size_t O_X1   = O_X0  + (size_t)TSTEPS * D0;
constexpr size_t O_X2   = O_X1  + (size_t)TSTEPS * D1;
constexpr size_t O_PEN  = O_X2  + (size_t)TSTEPS * D2;
constexpr size_t O_PXY  = O_PEN + (size_t)TSTEPS * 2;
constexpr size_t O_PDLP = O_PXY + (size_t)TSTEPS * NPIX * NPIX;

__device__ __forceinline__ unsigned short tobf_bits(float f) {
  unsigned u = __builtin_bit_cast(unsigned, f);
  unsigned r = u + 0x7FFFu + ((u >> 16) & 1u);   // round-to-nearest-even
  return (unsigned short)(r >> 16);
}

struct KParams {
  const float *W_a_0, *W_a_1, *b_a_0, *b_a_1, *tvec;
  const float *W_u_0, *W_u_1, *W_p, *b_p, *plv;
  const float *z1, *xp, *yp;
  const unsigned short *A1bf, *A2bf;   // bf16 bits
  const float *Abar0;
  float *xb0, *xb1, *xb2;              // [2][D] double-buffered state
  unsigned short *x1bf, *x2bf;         // bf16 copies of current x
  float *u1, *u2, *part1, *part2, *a1w, *a2w, *penw;
  unsigned *bar;
  float *out;
};

// ---- grid-wide sense/generation barrier (deterministic, atomic-based) ----
__device__ __forceinline__ void grid_sync(unsigned* bar, unsigned target) {
  __syncthreads();
  if (threadIdx.x == 0) {
    __threadfence();
    unsigned prev = atomicAdd(&bar[0], 1u);
    if (prev == NWG - 1u) {
      atomicExch(&bar[0], 0u);
      __threadfence();
      atomicExch(&bar[1], target);
    } else {
      volatile unsigned* g = &bar[1];
      while (*g < target) { __builtin_amdgcn_s_sleep(2); }
    }
    __threadfence();
  }
  __syncthreads();
}

// A tile 16x32 bf16, row-major source. Per ISA 16-bit A layout:
// lanes 0-15: row m=lane, slots0-7=K(j0..j0+7), slots8-15=K(j0+16..23)
// lanes16-31: row m=lane-16, slots0-7=K(j0+8..15), slots8-15=K(j0+24..31)
__device__ __forceinline__ v16bf load_a_tile(const unsigned short* A, int stride,
                                             int r0, int j0, int lane) {
  int row = r0 + (lane & 15);
  int kh  = lane >> 4;
  const unsigned short* p = A + (size_t)row * stride + j0 + 8 * kh;
  BFPack t;
  t.q[0] = *(const v4u*)(p);
  t.q[1] = *(const v4u*)(p + 16);
  __builtin_prefetch(p + 128, 0, 3);   // two unrolled iterations ahead, near-cache
  return t.v;
}

// B tile 32x16 with x replicated across all 16 columns:
// lanes 0-15 hold x[j0..j0+7],x[j0+16..j0+23]; lanes 16-31 hold x[j0+8..j0+15],x[j0+24..j0+31]
__device__ __forceinline__ v16bf load_b_vec(const unsigned short* xb, int j0, int lane) {
  const unsigned short* p = xb + j0 + 8 * (lane >> 4);
  BFPack t;
  t.q[0] = *(const v4u*)(p);
  t.q[1] = *(const v4u*)(p + 16);
  return t.v;
}

__device__ __forceinline__ void softmax16(const float* lg, float invT,
                                          float* dst, float* outp) {
  float m = -1e30f;
  for (int k = 0; k < 16; ++k) m = fmaxf(m, lg[k] * invT);
  float e[16], s = 0.f;
  for (int k = 0; k < 16; ++k) { e[k] = expf(lg[k] * invT - m); s += e[k]; }
  float inv = 1.f / s;
  for (int k = 0; k < 16; ++k) { float v = e[k] * inv; dst[k] = v; outp[k] = v; }
}

// ---- one-time prep kernels ----
__global__ void sketch_init(float* xb0, float* xb1, float* xb2, float* penw,
                            unsigned* bar, float* out, const float* z2,
                            const float* x0_1, const float* x0_2, const float* z1) {
  int gid = blockIdx.x * blockDim.x + threadIdx.x;
  int n = gridDim.x * blockDim.x;
  if (gid == 0) {
    bar[0] = 0u; bar[1] = 0u;
    penw[0] = 64.f; penw[1] = 64.f; penw[2] = 64.f; penw[3] = 64.f;
  }
  for (int i = gid; i < D0; i += n) xb0[i] = z2[i];     // carry x0 starts at z2
  for (int i = gid; i < D1; i += n) xb1[i] = x0_1[i];
  for (int i = gid; i < D2; i += n) xb2[i] = x0_2[i];
  for (size_t i = gid; i < (size_t)TSTEPS * 16; i += n) out[O_A0 + i] = z1[i & 15];
}

__global__ void sketch_convert(const float* A1, const float* A2,
                               unsigned short* A1b, unsigned short* A2b) {
  size_t n1 = (size_t)KMIX * D1 * D1, n2 = (size_t)KMIX * D2 * D2;
  size_t stride = (size_t)gridDim.x * blockDim.x;
  for (size_t i = blockIdx.x * (size_t)blockDim.x + threadIdx.x; i < n1 + n2; i += stride) {
    if (i < n1) A1b[i] = tobf_bits(A1[i]);
    else        A2b[i - n1] = tobf_bits(A2[i - n1]);
  }
}

__global__ void sketch_abar(const float* A0, const float* z1, float* Abar) {
  int idx = blockIdx.x * blockDim.x + threadIdx.x;
  if (idx < D0 * D0) {
    float s = 0.f;
#pragma unroll
    for (int k = 0; k < KMIX; ++k) s = fmaf(z1[k], A0[(size_t)k * D0 * D0 + idx], s);
    Abar[idx] = s;
  }
}

// ---- persistent scan kernel ----
__global__ void __launch_bounds__(NTHR, 1) sketch_run(KParams P) {
  __shared__ float lg1[16], lg2[16];
  __shared__ float s0[NTHR], s1[NTHR], s2[NTHR];
  __shared__ float shv[8];
  __shared__ float ex[NPIX], ey[NPIX];

  const int b = blockIdx.x;
  const int tid = threadIdx.x;
  const int lane = tid & 31;
  const int wave = __builtin_amdgcn_readfirstlane(b * (NTHR / 32) + (tid >> 5));
  int p = 0;
  unsigned barc = 0;

  const float invT1 = 1.f / P.tvec[1];
  const float invT2 = 1.f / P.tvec[2];

  for (int t = 0; t < TSTEPS; ++t) {
    const float* x0c = P.xb0 + p * D0;
    const float* x1c = P.xb1 + p * D1;
    const float* x2c = P.xb2 + p * D2;
    float* x0n = P.xb0 + (1 - p) * D0;
    float* x1n = P.xb1 + (1 - p) * D1;
    float* x2n = P.xb2 + (1 - p) * D2;

    // ---- Phase 1: u-vectors, attention logits+softmax, x0 update, x->bf16 ----
    if (b < 4) {
      int i = b * NTHR + tid;                       // 0..1023: u2 = W_u_1 @ x1
      const float* w = P.W_u_1 + (size_t)i * D1;
      float s = 0.f;
      for (int j = 0; j < D1; ++j) s = fmaf(w[j], x1c[j], s);
      P.u2[i] = s;
    } else if (b < 6) {
      int i = (b - 4) * NTHR + tid;                 // 0..511: u1 = W_u_0 @ x0
      const float* w = P.W_u_0 + (size_t)i * D0;
      float s = 0.f;
      for (int j = 0; j < D0; ++j) s = fmaf(w[j], x0c[j], s);
      P.u1[i] = s;
    } else if (b == 6) {
      int i = tid;                                  // x0n = x0 + Abar0@x0 * dt
      const float* w = P.Abar0 + (size_t)i * D0;
      float s = 0.f;
      for (int j = 0; j < D0; ++j) s = fmaf(w[j], x0c[j], s);
      float xv = x0c[i] + DTC * s;
      x0n[i] = xv;
      P.out[O_X0 + (size_t)t * D0 + i] = xv;
    } else if (b == 7) {
      if (tid < 16) {
        const float* w = P.W_a_0 + (size_t)tid * D0;
        float s = P.b_a_0[tid];
        for (int j = 0; j < D0; ++j) s = fmaf(w[j], x0c[j], s);
        lg1[tid] = s;
      } else if (tid < 32) {
        int r = tid - 16;
        const float* w = P.W_a_1 + (size_t)r * D1;
        float s = P.b_a_1[r];
        for (int j = 0; j < D1; ++j) s = fmaf(w[j], x1c[j], s);
        lg2[r] = s;
      }
      __syncthreads();
      if (tid == 0)      softmax16(lg1, invT1, P.a1w, P.out + O_A1o + (size_t)t * 16);
      else if (tid == 1) softmax16(lg2, invT2, P.a2w, P.out + O_A2o + (size_t)t * 16);
    } else if (b < 12) {
      int i = (b - 8) * NTHR + tid;                 // x2 -> bf16
      P.x2bf[i] = tobf_bits(x2c[i]);
    } else if (b < 14) {
      int i = (b - 12) * NTHR + tid;                // x1 -> bf16
      P.x1bf[i] = tobf_bits(x1c[i]);
    }
    grid_sync(P.bar, ++barc);

    // ---- Phase 2: WMMA mixture matvec partials (640 wave-tasks x 64 wmma) ----
    for (int task0 = wave; task0 < 640; task0 += NWAVES) {
      // force wave-uniform (scalar) task decode -> SGPR loop bounds & s_loads
      const int task = __builtin_amdgcn_readfirstlane(task0);
      const unsigned short *Am, *xbv;
      const float* aw;
      float* part;
      int D, r0, kbeg, kcnt, g;
      if (task < 512) {                              // A_2 mixture: 64 chunks x 8 k-pairs
        int chunk = task >> 3; g = task & 7;
        r0 = chunk * 16; D = D2; kbeg = g * 2; kcnt = 2;
        Am = P.A2bf; xbv = P.x2bf; aw = P.a2w; part = P.part2;
      } else {                                       // A_1 mixture: 32 chunks x 4 k-quads
        int tt = task - 512;
        int chunk = tt >> 2; g = tt & 3;
        r0 = chunk * 16; D = D1; kbeg = g * 4; kcnt = 4;
        Am = P.A1bf; xbv = P.x1bf; aw = P.a1w; part = P.part1;
      }
      v8f acc = {0.f, 0.f, 0.f, 0.f, 0.f, 0.f, 0.f, 0.f};
      for (int kk = 0; kk < kcnt; ++kk) {
        const int k = kbeg + kk;
        const unsigned short* Ak = Am + (size_t)k * D * D;
        const float wgt = aw[k];                     // scalar (s_load), hoisted
        v8f c = {0.f, 0.f, 0.f, 0.f, 0.f, 0.f, 0.f, 0.f};
        for (int j0 = 0; j0 < D; j0 += 64) {         // unroll x2: overlap loads & XDL
          v16bf av0 = load_a_tile(Ak, D, r0, j0, lane);
          v16bf bv0 = load_b_vec(xbv, j0, lane);
          v16bf av1 = load_a_tile(Ak, D, r0, j0 + 32, lane);
          v16bf bv1 = load_b_vec(xbv, j0 + 32, lane);
          c = __builtin_amdgcn_wmma_f32_16x16x32_bf16(false, av0, false, bv0,
                                                      (short)0, c, false, false);
          c = __builtin_amdgcn_wmma_f32_16x16x32_bf16(false, av1, false, bv1,
                                                      (short)0, c, false, false);
        }
        acc += c * wgt;
      }
      if ((lane & 15) == 0) {                        // column 0 carries the matvec
        int rb = r0 + 8 * (lane >> 4);
#pragma unroll
        for (int v = 0; v < 8; ++v) part[(size_t)g * D + rb + v] = acc[v];
      }
    }
    grid_sync(P.bar, ++barc);

    // ---- Phase 3: fixed-order reduction, state update, x outputs ----
    {
      int gid = b * NTHR + tid;
      if (gid < D2) {
        float s = P.u2[gid];
#pragma unroll
        for (int g = 0; g < 8; ++g) s += P.part2[(size_t)g * D2 + gid];
        float xv = x2c[gid] + DTC * s;
        x2n[gid] = xv;
        P.out[O_X2 + (size_t)t * D2 + gid] = xv;
      } else if (gid < D2 + D1) {
        int i = gid - D2;
        float s = P.u1[i];
#pragma unroll
        for (int g = 0; g < 4; ++g) s += P.part1[(size_t)g * D1 + i];
        float xv = x1c[i] + DTC * s;
        x1n[i] = xv;
        P.out[O_X1 + (size_t)t * D1 + i] = xv;
      }
    }
    grid_sync(P.bar, ++barc);

    // ---- Phase 4: pen head + outer-product image (redundant per WG) ----
    {
      float p0 = 0.f, p1 = 0.f, p2 = 0.f;
#pragma unroll
      for (int q = 0; q < 4; ++q) {
        int i = tid + q * NTHR;
        float xv = x2n[i];
        p0 = fmaf(P.W_p[i], xv, p0);
        p1 = fmaf(P.W_p[D2 + i], xv, p1);
        p2 = fmaf(P.W_p[2 * D2 + i], xv, p2);
      }
      s0[tid] = p0; s1[tid] = p1; s2[tid] = p2;
      __syncthreads();
      if (tid < 3) {
        const float* arr = (tid == 0) ? s0 : ((tid == 1) ? s1 : s2);
        float s = P.b_p[tid];
        for (int j = 0; j < NTHR; ++j) s += arr[j];   // fixed order -> deterministic
        shv[tid] = s;
      }
      __syncthreads();
      if (tid == 0) {
        float pz   = shv[2];
        float sig  = 1.f / (1.f + expf(-pz));          // exp(log_sigmoid) == sigmoid
        float pdlp = (pz >= 0.f) ? -log1pf(expf(-pz)) : (pz - log1pf(expf(pz)));
        float ivar = 1.f / (expf(P.plv[0]) + 0.001f);
        float pen0 = P.penw[p * 2 + 0], pen1 = P.penw[p * 2 + 1];
        float pn0 = 128.f / (1.f + expf(-((pen0 + shv[0] - 64.f) * 0.078125f)));
        float pn1 = 128.f / (1.f + expf(-((pen1 + shv[1] - 64.f) * 0.078125f)));
        shv[3] = pen0; shv[4] = pen1; shv[5] = ivar; shv[6] = sig; shv[7] = pdlp;
        if (b == 0) {
          P.penw[(1 - p) * 2 + 0] = pn0;
          P.penw[(1 - p) * 2 + 1] = pn1;
          P.out[O_PEN + (size_t)t * 2 + 0] = pn0;
          P.out[O_PEN + (size_t)t * 2 + 1] = pn1;
          P.out[O_PDLP + t] = pdlp;
        }
      }
      __syncthreads();
      if (tid < NPIX) {
        float d = shv[3] - P.xp[tid];
        ex[tid] = expf(-0.5f * d * d * shv[5]);
      } else {
        int r = tid - NPIX;
        float d = shv[4] - P.yp[r];
        ey[r] = shv[6] * expf(-0.5f * d * d * shv[5]);
      }
      __syncthreads();
      int idx = b * NTHR + tid;                       // 0..16383, 256 per WG
      int r = idx >> 7, c = idx & 127;
      P.out[O_PXY + (size_t)t * (NPIX * NPIX) + idx] = ey[r] * ex[c];
      __syncthreads();
    }
    p ^= 1;
  }
}

extern "C" void kernel_launch(void* const* d_in, const int* in_sizes, int n_in,
                              void* d_out, int out_size, void* d_ws, size_t ws_size,
                              hipStream_t stream) {
  const float* W_a_0 = (const float*)d_in[0];
  const float* W_a_1 = (const float*)d_in[1];
  const float* b_a_0 = (const float*)d_in[2];
  const float* b_a_1 = (const float*)d_in[3];
  const float* tvec  = (const float*)d_in[4];
  const float* W_u_0 = (const float*)d_in[5];
  const float* W_u_1 = (const float*)d_in[6];
  const float* W_p   = (const float*)d_in[7];
  const float* b_p   = (const float*)d_in[8];
  const float* plv   = (const float*)d_in[9];
  const float* A_0   = (const float*)d_in[10];
  const float* A_1   = (const float*)d_in[11];
  const float* A_2   = (const float*)d_in[12];
  const float* x0_1  = (const float*)d_in[14];
  const float* x0_2  = (const float*)d_in[15];
  const float* z1    = (const float*)d_in[16];
  const float* z2    = (const float*)d_in[17];
  const float* xp    = (const float*)d_in[18];
  const float* yp    = (const float*)d_in[19];

  char* ws = (char*)d_ws;
  size_t off = 0;
  auto take = [&](size_t bytes) -> char* {
    char* r = ws + off;
    off = (off + bytes + 255) & ~(size_t)255;
    return r;
  };
  unsigned short* A2bf = (unsigned short*)take((size_t)KMIX * D2 * D2 * 2);
  unsigned short* A1bf = (unsigned short*)take((size_t)KMIX * D1 * D1 * 2);
  float* Abar0 = (float*)take((size_t)D0 * D0 * 4);
  float* xb0   = (float*)take(2 * D0 * 4);
  float* xb1   = (float*)take(2 * D1 * 4);
  float* xb2   = (float*)take(2 * D2 * 4);
  unsigned short* x2bf = (unsigned short*)take(D2 * 2);
  unsigned short* x1bf = (unsigned short*)take(D1 * 2);
  float* u1    = (float*)take(D1 * 4);
  float* u2    = (float*)take(D2 * 4);
  float* part1 = (float*)take(4 * D1 * 4);
  float* part2 = (float*)take(8 * D2 * 4);
  float* a1w   = (float*)take(16 * 4);
  float* a2w   = (float*)take(16 * 4);
  float* penw  = (float*)take(4 * 4);
  unsigned* bar = (unsigned*)take(2 * 4);

  float* out = (float*)d_out;

  sketch_init<<<64, 256, 0, stream>>>(xb0, xb1, xb2, penw, bar, out, z2, x0_1, x0_2, z1);
  sketch_convert<<<4096, 256, 0, stream>>>(A_1, A_2, A1bf, A2bf);
  sketch_abar<<<(D0 * D0) / 256, 256, 0, stream>>>(A_0, z1, Abar0);

  KParams P;
  P.W_a_0 = W_a_0; P.W_a_1 = W_a_1; P.b_a_0 = b_a_0; P.b_a_1 = b_a_1; P.tvec = tvec;
  P.W_u_0 = W_u_0; P.W_u_1 = W_u_1; P.W_p = W_p; P.b_p = b_p; P.plv = plv;
  P.z1 = z1; P.xp = xp; P.yp = yp;
  P.A1bf = A1bf; P.A2bf = A2bf; P.Abar0 = Abar0;
  P.xb0 = xb0; P.xb1 = xb1; P.xb2 = xb2;
  P.x1bf = x1bf; P.x2bf = x2bf;
  P.u1 = u1; P.u2 = u2; P.part1 = part1; P.part2 = part2;
  P.a1w = a1w; P.a2w = a2w; P.penw = penw; P.bar = bar;
  P.out = out;

  sketch_run<<<NWG, NTHR, 0, stream>>>(P);
}
